// BinaryDense_52836687675990
// MI455X (gfx1250) — matrix operand order repaired and checked by reading, not compiled, not run on hardware
//
#include <hip/hip_runtime.h>

// ---------------------------------------------------------------------------
// BinaryDense for MI455X (gfx1250): out = x @ binarize(W) + bias
//   M=2097152, K=128, N=128, fp32 in/out.
// Memory-bound: ~2.15 GB total traffic @ 23.3 TB/s => ~92us floor. Compute is
// done on the bf16 WMMA pipe with an x = hi + lo bf16 split (w_bin = +/-1 is
// exact in bf16), giving ~fp32 accuracy while the 137 GFLOP of bf16 WMMA work
// stays far below the HBM floor. fp32 WMMA (16x16x4) would be compute-bound;
// avoided.
// NOTE: stochastic binarization uses a device hash RNG; it cannot bit-match
// jax.random.bernoulli(threefry, key(42)).
// ---------------------------------------------------------------------------

typedef __attribute__((ext_vector_type(16))) __bf16 v16bf;
typedef __attribute__((ext_vector_type(8)))  float  v8f;

#define M_DIM 2097152
#define K_DIM 128
#define N_DIM 128
#define ROWS  32    // rows staged per workgroup iteration (2 m-tiles)
#define PITCH 136   // bf16 elems per LDS row: 128 + 8 pad (272B = 68 dwords,
                    // 16B aligned; rows land on distinct banks for frag loads)

__device__ __forceinline__ unsigned pcg_hash(unsigned v) {
    unsigned s = v * 747796405u + 2891336453u;
    unsigned w = ((s >> ((s >> 28) + 4u)) ^ s) * 277803737u;
    return (w >> 22) ^ w;
}

// ---------------------------------------------------------------------------
// Kernel 1: binarize W (K x N row-major) and emit it pre-swizzled in the
// bf16 B-fragment order used here, k contiguous per lane half (32x16 tile, wave32):
//   lane l (0..15):  column n = nt*16 + l,      K = kt*32 + 0..15
//   lane l (16..31): column n = nt*16 + (l-16), K = kt*32 + 16..31
// Fragment f = nt*4 + kt; per (frag,lane): 16 bf16 = 32 contiguous bytes at
// wpack + (f*32 + lane)*16. Total 32 KB in d_ws.
// ---------------------------------------------------------------------------
__global__ void __launch_bounds__(256)
binarize_pack_kernel(const float* __restrict__ weight,
                     const int* __restrict__ is_training,
                     __bf16* __restrict__ wpack) {
    const int tid  = blockIdx.x * blockDim.x + threadIdx.x;  // 0..1023
    const int lane = tid & 31;
    const int frag = tid >> 5;        // 0..31
    const int nt   = frag >> 2;       // n-tile 0..7
    const int kt   = frag & 3;        // k-tile 0..3
    const int n     = nt * 16 + (lane & 15);
    const int kbase = kt * 32 + ((lane >> 4) ? 16 : 0);
    const int train = is_training[0];

    __bf16 vals[16];
#pragma unroll
    for (int i = 0; i < 16; ++i) {
        const int k   = kbase + i;
        const int idx = k * N_DIM + n;
        const float wv = weight[idx];
        float r;
        if (train) {
            // Bernoulli(hard_sigmoid(w)) via counter-based hash
            float prob = fminf(fmaxf((wv + 1.0f) * 0.5f, 0.0f), 1.0f);
            unsigned h = pcg_hash((unsigned)idx ^ 0x2545F491u);
            float u = (float)h * 2.3283064365386963e-10f;  // [0,1)
            r = (u < prob) ? 1.0f : -1.0f;
        } else {
            r = (wv > 0.0f) ? 1.0f : -1.0f;
        }
        vals[i] = (__bf16)r;
    }
    uint4 p0, p1;
    __builtin_memcpy(&p0, &vals[0], 16);
    __builtin_memcpy(&p1, &vals[8], 16);
    uint4* dst = (uint4*)(wpack + (size_t)tid * 16);
    dst[0] = p0;
    dst[1] = p1;
}

// ---------------------------------------------------------------------------
// Kernel 2: streaming GEMM. 256 threads = 8 wave32.
// Wave w owns n-tile w (cols w*16..w*16+15) and keeps its 4 B fragments
// (full K=128) resident in 32 VGPRs for the entire grid-stride loop.
// Per 32-row block:
//   - cooperative load of 32x128 fp32 x, split into bf16 hi/lo, staged in LDS
//   - each wave, per m-tile (x2): preload ALL 8 A fragments (hi/lo x 4 kt)
//     into registers first (one ds-wait), then run 8 back-to-back
//     v_wmma_f32_16x16x32_bf16 on two independent accumulator chains
//   - coalesced fp32 stores of each 16x16 tile (+bias)
// ---------------------------------------------------------------------------
__global__ void __launch_bounds__(256)
binary_dense_gemm(const float* __restrict__ x,
                  const float* __restrict__ bias,
                  const __bf16* __restrict__ wpack,
                  float* __restrict__ out) {
    __shared__ __bf16 ldsHi[ROWS * PITCH];
    __shared__ __bf16 ldsLo[ROWS * PITCH];

    const int tid  = threadIdx.x;
    const int wave = tid >> 5;   // 0..7  -> n-tile
    const int lane = tid & 31;

    // Persistent B fragments for this wave's n-tile (pre-swizzled in d_ws).
    v16bf Bf[4];
#pragma unroll
    for (int kt = 0; kt < 4; ++kt) {
        const uint4* src =
            (const uint4*)(wpack + (size_t)((wave * 4 + kt) * 32 + lane) * 16);
        ((uint4*)&Bf[kt])[0] = src[0];
        ((uint4*)&Bf[kt])[1] = src[1];
    }

    const float bv = bias[wave * 16 + (lane & 15)];

    // Cooperative x loader mapping: thread -> (row, 16 consecutive cols)
    const int row  = tid >> 3;           // 0..31
    const int col0 = (tid & 7) * 16;     // 0,16,...,112

    // A-fragment gather mapping (contiguous k per lane half, same as B pack):
    //   lanes 0..15:  M=lane,     K = kt*32 + 0..15
    //   lanes 16..31: M=lane-16,  K = kt*32 + 16..31
    const int aRow  = lane & 15;
    const int aHalf = lane >> 4;
    const int aOff  = aRow * PITCH + aHalf * 16;

    const int nblocks = M_DIM / ROWS;
    for (int rb = blockIdx.x; rb < nblocks; rb += gridDim.x) {
        // ---- stage 32x128 x block into LDS as bf16 hi + lo -------------
        const float* xr = x + (size_t)(rb * ROWS + row) * K_DIM + col0;
        float4 f0 = *(const float4*)(xr);
        float4 f1 = *(const float4*)(xr + 4);
        float4 f2 = *(const float4*)(xr + 8);
        float4 f3 = *(const float4*)(xr + 12);
        float fv[16] = {f0.x, f0.y, f0.z, f0.w, f1.x, f1.y, f1.z, f1.w,
                        f2.x, f2.y, f2.z, f2.w, f3.x, f3.y, f3.z, f3.w};
        __bf16 hi[16], lo[16];
#pragma unroll
        for (int i = 0; i < 16; ++i) {
            __bf16 h = (__bf16)fv[i];
            hi[i] = h;
            lo[i] = (__bf16)(fv[i] - (float)h);
        }
        __builtin_memcpy(&ldsHi[row * PITCH + col0], hi, 32);  // 2x ds_store_b128
        __builtin_memcpy(&ldsLo[row * PITCH + col0], lo, 32);

        // speculative prefetch of next row block -> global_prefetch_b8
        __builtin_prefetch((const void*)(xr + (size_t)gridDim.x * ROWS * K_DIM), 0, 1);

        __syncthreads();

        // ---- two 16-row m-tiles per wave -------------------------------
#pragma unroll
        for (int mt = 0; mt < 2; ++mt) {
            const int base = mt * 16 * PITCH + aOff;

            // Preload all 8 A fragments (one ds-wait covers the clause).
            v16bf Ah[4], Al[4];
#pragma unroll
            for (int kt = 0; kt < 4; ++kt) {
                const int o1 = base + kt * 32;
                ((uint4*)&Ah[kt])[0] = *(const uint4*)&ldsHi[o1];
                ((uint4*)&Ah[kt])[1] = *(const uint4*)&ldsHi[o1 + 8];
                ((uint4*)&Al[kt])[0] = *(const uint4*)&ldsLo[o1];
                ((uint4*)&Al[kt])[1] = *(const uint4*)&ldsLo[o1 + 8];
            }

            // 8 WMMAs, two independent accumulation chains (hi / lo).
            v8f c0 = {};
            v8f c1 = {};
#pragma unroll
            for (int kt = 0; kt < 4; ++kt) {
                c0 = __builtin_amdgcn_wmma_f32_16x16x32_bf16(
                    false, Ah[kt], false, Bf[kt], (short)0, c0, false, false);
                c1 = __builtin_amdgcn_wmma_f32_16x16x32_bf16(
                    false, Al[kt], false, Bf[kt], (short)0, c1, false, false);
            }

            // ---- write 16x16 fp32 tile (+bias) -------------------------
            // C layout: VGPR j: lanes 0..15 -> M=j, lanes 16..31 -> M=j+8;
            //           N = lane%16
            float* orow = out + ((size_t)rb * ROWS + mt * 16) * N_DIM
                              + wave * 16 + (lane & 15);
            const int mbase = (lane >> 4) ? 8 : 0;
#pragma unroll
            for (int j = 0; j < 8; ++j) {
                orow[(size_t)(mbase + j) * N_DIM] = c0[j] + c1[j] + bv;
            }
        }

        __syncthreads();  // before next iteration overwrites LDS
    }
}

extern "C" void kernel_launch(void* const* d_in, const int* in_sizes, int n_in,
                              void* d_out, int out_size, void* d_ws, size_t ws_size,
                              hipStream_t stream) {
    (void)in_sizes; (void)n_in; (void)out_size; (void)ws_size;
    const float* x           = (const float*)d_in[0];
    const float* weight      = (const float*)d_in[1];
    const float* bias        = (const float*)d_in[2];
    const int*   is_training = (const int*)d_in[3];
    float*       out         = (float*)d_out;
    __bf16*      wpack       = (__bf16*)d_ws;   // 32 KB packed B fragments

    binarize_pack_kernel<<<4, 256, 0, stream>>>(weight, is_training, wpack);
    // 2048 WGs x 8 waves; each WG handles 65536/2048 = 32 row blocks.
    binary_dense_gemm<<<2048, 256, 0, stream>>>(x, bias, wpack, out);
}
